// INT8MultiHeadSelfAttention_76596446757280
// MI455X (gfx1250) — 16-bit and 8-bit matrix operand order repaired and checked by reading, compiled, not run on hardware
//
#include <hip/hip_runtime.h>
#include <hip/hip_fp16.h>
#include <cmath>

#define HIDDEN   768
#define SEQ      1024
#define BATCH    4
#define NHEADS   12
#define HEADDIM  64
#define NTOK     (BATCH*SEQ)
#define QMAXF    127.0f

typedef __attribute__((ext_vector_type(16))) _Float16 v16h;
typedef __attribute__((ext_vector_type(8)))  float    v8f;
typedef __attribute__((ext_vector_type(8)))  int      v8i;

typedef unsigned int u32x4 __attribute__((ext_vector_type(4)));
typedef int          i32x4 __attribute__((ext_vector_type(4)));
typedef int          i32x8 __attribute__((ext_vector_type(8)));

union F16Frag { v16h v; float4 q[2]; };
union I8Frag  { v8i  v; float2 d[4]; float4 q[2]; };

#ifndef __has_builtin
#define __has_builtin(x) 0
#endif
#if __has_builtin(__builtin_amdgcn_tensor_load_to_lds) && \
    __has_builtin(__builtin_amdgcn_s_wait_tensorcnt)
#define USE_TDM 1
#else
#define USE_TDM 0
#endif

#if USE_TDM
__device__ __forceinline__ void tdm_load_tile_2d(
    unsigned lds_addr, const void* gptr,
    unsigned tile_w, unsigned tile_h,
    unsigned tensor_w, unsigned tensor_h,
    unsigned long long row_stride_elems) {
  const unsigned long long ga = (unsigned long long)(uintptr_t)gptr;
  u32x4 g0;
  g0.x = 1u;
  g0.y = lds_addr;
  g0.z = (unsigned)(ga & 0xffffffffu);
  g0.w = (unsigned)((ga >> 32) & 0x1ffffffu)
       | (2u << 30);
  i32x8 g1;
  g1[0] = 0;
  g1[1] = (int)((tensor_w & 0xffffu) << 16);
  g1[2] = (int)((tensor_w >> 16) | ((tensor_h & 0xffffu) << 16));
  g1[3] = (int)((tensor_h >> 16) | (tile_w << 16));
  g1[4] = (int)(tile_h & 0xffffu);
  g1[5] = (int)(row_stride_elems & 0xffffffffu);
  g1[6] = (int)((row_stride_elems >> 32) & 0xffffu);
  g1[7] = 0;
  const i32x4 z4 = {0, 0, 0, 0};
#if __clang_major__ >= 23
  const i32x8 z8 = {0, 0, 0, 0, 0, 0, 0, 0};
  __builtin_amdgcn_tensor_load_to_lds(g0, g1, z4, z4, z8, 0);
#else
  __builtin_amdgcn_tensor_load_to_lds(g0, g1, z4, z4, 0);
#endif
}
#endif

__global__ void pack_int8_kernel(const int* __restrict__ w,
                                 unsigned char* __restrict__ out, int n4) {
  int i = blockIdx.x * blockDim.x + threadIdx.x;
  if (i < n4) {
    int j = i * 4;
    unsigned int p = (unsigned int)(w[j] & 0xff)
                   | ((unsigned int)(w[j + 1] & 0xff) << 8)
                   | ((unsigned int)(w[j + 2] & 0xff) << 16)
                   | ((unsigned int)(w[j + 3] & 0xff) << 24);
    ((unsigned int*)out)[i] = p;
  }
}

__global__ __launch_bounds__(256) void quant_rows_kernel(
    const float* __restrict__ x, signed char* __restrict__ xq,
    float* __restrict__ scales, int nrows) {
  const int wave = threadIdx.x >> 5, lane = threadIdx.x & 31;
  const int row = blockIdx.x * 8 + wave;
  if (row >= nrows) return;
  const float4* xr4 = (const float4*)(x + (size_t)row * HIDDEN);
  float4 vals[6];
  float amax = 0.f;
  #pragma unroll
  for (int j = 0; j < 6; ++j) {
    vals[j] = xr4[lane + 32 * j];
    amax = fmaxf(amax, fmaxf(fmaxf(fabsf(vals[j].x), fabsf(vals[j].y)),
                             fmaxf(fabsf(vals[j].z), fabsf(vals[j].w))));
  }
  for (int off = 16; off > 0; off >>= 1)
    amax = fmaxf(amax, __shfl_xor(amax, off, 32));
  float scale = amax / QMAXF;
  if (scale == 0.f) scale = 1.f;
  const float inv = 1.f / scale;
  if (lane == 0) scales[row] = scale;
  unsigned int* oq4 = (unsigned int*)(xq + (size_t)row * HIDDEN);
  #pragma unroll
  for (int j = 0; j < 6; ++j) {
    int qa = (int)fminf(fmaxf(rintf(vals[j].x * inv), -128.f), 127.f);
    int qb = (int)fminf(fmaxf(rintf(vals[j].y * inv), -128.f), 127.f);
    int qc = (int)fminf(fmaxf(rintf(vals[j].z * inv), -128.f), 127.f);
    int qd = (int)fminf(fmaxf(rintf(vals[j].w * inv), -128.f), 127.f);
    oq4[lane + 32 * j] = (unsigned)(qa & 0xff) | ((unsigned)(qb & 0xff) << 8)
                       | ((unsigned)(qc & 0xff) << 16) | ((unsigned)(qd & 0xff) << 24);
  }
}

template <int MODE>
__global__ __launch_bounds__(256) void gemm_i8_kernel(
    const signed char* __restrict__ Xq, const float* __restrict__ xscale,
    const signed char* __restrict__ W,  const float* __restrict__ wscale,
    const float* __restrict__ bias, void* __restrict__ dst) {
  __shared__ signed char sA[128 * 64];
  __shared__ signed char sB[64 * 64];
  const int tid  = threadIdx.x;
  const int wave = tid >> 5, lane = tid & 31;
  const int hi = lane >> 4, lc = lane & 15;
  const int m0 = blockIdx.y * 128;
  const int n0 = blockIdx.x * 64;

  const v8i zi = {0, 0, 0, 0, 0, 0, 0, 0};
  v8i acc[4];
  acc[0] = zi; acc[1] = zi; acc[2] = zi; acc[3] = zi;

  for (int ks = 0; ks < HIDDEN / 64; ++ks) {
    const int kb = ks * 64;
    __syncthreads();
#if USE_TDM
    if (wave == 0) {
      tdm_load_tile_2d((unsigned)(uintptr_t)&sA[0],
                       &Xq[(size_t)m0 * HIDDEN + kb],
                       64u, 128u, (unsigned)HIDDEN, (unsigned)NTOK,
                       (unsigned long long)HIDDEN);
      tdm_load_tile_2d((unsigned)(uintptr_t)&sB[0],
                       &W[(size_t)n0 * HIDDEN + kb],
                       64u, 64u, (unsigned)HIDDEN, (unsigned)HIDDEN,
                       (unsigned long long)HIDDEN);
      __builtin_amdgcn_s_wait_tensorcnt(0);
    }
#else
    #pragma unroll
    for (int j = 0; j < 2; ++j) {
      int c = tid + j * 256;
      int r = c >> 2, cc = (c & 3) * 16;
      *(float4*)&sA[r * 64 + cc] =
          *(const float4*)&Xq[(size_t)(m0 + r) * HIDDEN + kb + cc];
    }
    {
      int r = tid >> 2, cc = (tid & 3) * 16;
      *(float4*)&sB[r * 64 + cc] =
          *(const float4*)&W[(size_t)(n0 + r) * HIDDEN + kb + cc];
    }
#endif
    if (ks + 1 < HIDDEN / 64) {
      __builtin_prefetch(&Xq[(size_t)(m0 + (tid >> 1)) * HIDDEN + kb + 64], 0, 1);
      __builtin_prefetch(&W[(size_t)(n0 + (tid >> 2)) * HIDDEN + kb + 64], 0, 1);
    }
    __syncthreads();

    I8Frag a;
    const signed char* ar = &sA[(wave * 16 + lc) * 64 + hi * 16];
    a.q[0] = *(const float4*)(ar + 0);
    a.q[1] = *(const float4*)(ar + 32);
    #pragma unroll
    for (int nt = 0; nt < 4; ++nt) {
      I8Frag bf;
      const signed char* br = &sB[(nt * 16 + lc) * 64 + hi * 16];
      bf.q[0] = *(const float4*)(br + 0);
      bf.q[1] = *(const float4*)(br + 32);
      acc[nt] = __builtin_amdgcn_wmma_i32_16x16x64_iu8(
          true, a.v, true, bf.v, acc[nt], false, false);
    }
  }

  #pragma unroll
  for (int nt = 0; nt < 4; ++nt) {
    const int n = n0 + nt * 16 + lc;
    const float ws = wscale[n], bs = bias[n];
    #pragma unroll
    for (int v = 0; v < 8; ++v) {
      const int m = m0 + wave * 16 + v + hi * 8;
      const float val = (float)acc[nt][v] * ws * xscale[m] + bs;
      if (MODE == 0) {
        const int b = m >> 10, s = m & 1023;
        const int h = n >> 6, d = n & 63;
        ((__half*)dst)[(((size_t)(b * NHEADS + h)) * SEQ + s) * HEADDIM + d] =
            __float2half(val);
      } else {
        ((float*)dst)[(size_t)m * HIDDEN + n] = val;
      }
    }
  }
}

__global__ void pla_coeff_kernel(float* __restrict__ coef) {
  if (threadIdx.x != 0 || blockIdx.x != 0) return;
  for (int i = 0; i < 12; ++i) {
    const double l = -10.0 + 10.0 * i / 12.0;
    const double r = -10.0 + 10.0 * (i + 1) / 12.0;
    double n = 0, sx = 0, sy = 0, sxx = 0, sxy = 0;
    for (int j = 0; j <= 1000; ++j) {
      const double x = -10.0 + 10.0 * j / 1000.0;
      if (x >= l && x <= r) {
        const double y = exp(x);
        n += 1.0; sx += x; sy += y; sxx += x * x; sxy += x * y;
      }
    }
    const double m = (n * sxy - sx * sy) / (n * sxx - sx * sx);
    const double c = (sy - m * sx) / n;
    coef[2 * i] = (float)m;
    coef[2 * i + 1] = (float)c;
  }
}

__global__ __launch_bounds__(128) void attn_kernel(
    const __half* __restrict__ Q, const __half* __restrict__ K,
    const __half* __restrict__ V, const float* __restrict__ mask,
    const float* __restrict__ coef, float* __restrict__ ctx) {
  __shared__ __half sS[64][SEQ];
  __shared__ __half sVt[64][32];
  __shared__ float  sInv[64];
  __shared__ float  sCoef[24];

  const int tid = threadIdx.x;
  const int wave = tid >> 5, lane = tid & 31;
  const int hi = lane >> 4, lc = lane & 15;
  const int b = blockIdx.z, h = blockIdx.y;
  const int q0 = blockIdx.x * 64;
  const int bh = b * NHEADS + h;
  const float scale = 0.125f;
  const __half* Qb = Q + (size_t)bh * SEQ * HEADDIM;
  const __half* Kb = K + (size_t)bh * SEQ * HEADDIM;
  const __half* Vb = V + (size_t)bh * SEQ * HEADDIM;

  if (tid < 24) sCoef[tid] = coef[tid];

  F16Frag qa[2];
  {
    const __half* qr = Qb + (size_t)(q0 + wave * 16 + lc) * HEADDIM;
    #pragma unroll
    for (int kh = 0; kh < 2; ++kh) {
      qa[kh].q[0] = *(const float4*)(qr + kh * 32 + hi * 8);
      qa[kh].q[1] = *(const float4*)(qr + kh * 32 + 16 + hi * 8);
    }
  }

  const float* mrow = mask + (size_t)b * SEQ;
  for (int kt = 0; kt < SEQ / 16; ++kt) {
    const int kcol = kt * 16 + lc;
    const __half* kr = Kb + (size_t)kcol * HEADDIM;
    v8f acc = {0.f, 0.f, 0.f, 0.f, 0.f, 0.f, 0.f, 0.f};
    #pragma unroll
    for (int kh = 0; kh < 2; ++kh) {
      F16Frag bf;
      bf.q[0] = *(const float4*)(kr + kh * 32 + hi * 8);
      bf.q[1] = *(const float4*)(kr + kh * 32 + 16 + hi * 8);
      acc = __builtin_amdgcn_wmma_f32_16x16x32_f16(
          false, qa[kh].v, false, bf.v, (short)0, acc, false, false);
    }
    const float mk = mrow[kcol];
    #pragma unroll
    for (int v = 0; v < 8; ++v)
      sS[wave * 16 + v + hi * 8][kcol] = __float2half(acc[v] * scale + mk);
  }
  __syncthreads();

  {
    const int rl = wave * 16 + (lane >> 1);
    const int c0 = (lane & 1) * 512;
    union HV { float4 f4; __half hh[8]; } u;
    float mx = -3.0e38f;
    for (int c = 0; c < 512; c += 8) {
      u.f4 = *(const float4*)&sS[rl][c0 + c];
      #pragma unroll
      for (int j = 0; j < 8; ++j) mx = fmaxf(mx, __half2float(u.hh[j]));
    }
    mx = fmaxf(mx, __shfl_xor(mx, 1, 32));
    float sum = 0.f;
    for (int c = 0; c < 512; c += 8) {
      u.f4 = *(const float4*)&sS[rl][c0 + c];
      #pragma unroll
      for (int j = 0; j < 8; ++j) {
        float x = __half2float(u.hh[j]) - mx;
        float xc = fminf(fmaxf(x, -10.f), 0.f);
        int idx = (int)((xc + 10.f) * 1.2f);
        idx = idx < 0 ? 0 : (idx > 10 ? 10 : idx);
        float e = sCoef[2 * idx] * xc + sCoef[2 * idx + 1];
        u.hh[j] = __float2half(e);
        sum += e;
      }
      *(float4*)&sS[rl][c0 + c] = u.f4;
    }
    sum += __shfl_xor(sum, 1, 32);
    if ((lane & 1) == 0) sInv[rl] = 1.f / (sum + 1e-9f);
  }

  const v8f zf = {0.f, 0.f, 0.f, 0.f, 0.f, 0.f, 0.f, 0.f};
  v8f cacc[4];
  cacc[0] = zf; cacc[1] = zf; cacc[2] = zf; cacc[3] = zf;
  for (int kc = 0; kc < SEQ / 32; ++kc) {
    __syncthreads();
    {
      const int kk = tid >> 2, dseg = (tid & 3) * 16;
      const __half* vsrc = Vb + (size_t)(kc * 32 + kk) * HEADDIM + dseg;
      #pragma unroll
      for (int j = 0; j < 16; ++j) sVt[dseg + j][kk] = vsrc[j];
    }
    __syncthreads();
    F16Frag pa;
    {
      const __half* pr = &sS[wave * 16 + lc][kc * 32];
      pa.q[0] = *(const float4*)(pr + hi * 8);
      pa.q[1] = *(const float4*)(pr + 16 + hi * 8);
    }
    #pragma unroll
    for (int dt = 0; dt < 4; ++dt) {
      F16Frag vf;
      const __half* vr = &sVt[dt * 16 + lc][hi * 8];
      vf.q[0] = *(const float4*)(vr + 0);
      vf.q[1] = *(const float4*)(vr + 16);
      cacc[dt] = __builtin_amdgcn_wmma_f32_16x16x32_f16(
          false, pa.v, false, vf.v, (short)0, cacc[dt], false, false);
    }
  }

  #pragma unroll
  for (int dt = 0; dt < 4; ++dt) {
    const int d = dt * 16 + lc;
    #pragma unroll
    for (int v = 0; v < 8; ++v) {
      const int rl = wave * 16 + v + hi * 8;
      const float val = cacc[dt][v] * sInv[rl];
      const int s = q0 + rl;
      ctx[((size_t)(b * SEQ + s)) * HIDDEN + h * HEADDIM + d] = val;
    }
  }
}

extern "C" void kernel_launch(void* const* d_in, const int* in_sizes, int n_in,
                              void* d_out, int out_size, void* d_ws,
                              size_t ws_size, hipStream_t stream) {
  (void)in_sizes; (void)n_in; (void)out_size; (void)ws_size;
  const float* hidden = (const float*)d_in[0];
  const float* mask   = (const float*)d_in[1];
  const int*   w_i32[4]  = {(const int*)d_in[2],  (const int*)d_in[5],
                            (const int*)d_in[8],  (const int*)d_in[11]};
  const float* w_sc[4]   = {(const float*)d_in[3], (const float*)d_in[6],
                            (const float*)d_in[9], (const float*)d_in[12]};
  const float* w_bias[4] = {(const float*)d_in[4], (const float*)d_in[7],
                            (const float*)d_in[10], (const float*)d_in[13]};

  char* ws = (char*)d_ws;
  size_t off = 0;
  auto carve = [&](size_t bytes) -> char* {
    char* p = ws + off;
    off += (bytes + 255) & ~(size_t)255;
    return p;
  };
  signed char* Xq = (signed char*)carve((size_t)NTOK * HIDDEN);
  float* xscale = (float*)carve((size_t)NTOK * 4);
  signed char* Wp[4];
  for (int i = 0; i < 4; ++i) Wp[i] = (signed char*)carve((size_t)HIDDEN * HIDDEN);
  __half* qkv[3];
  for (int i = 0; i < 3; ++i) qkv[i] = (__half*)carve((size_t)NTOK * HIDDEN * 2);
  float* ctx = (float*)carve((size_t)NTOK * HIDDEN * 4);
  signed char* Cq = (signed char*)carve((size_t)NTOK * HIDDEN);
  float* cscale = (float*)carve((size_t)NTOK * 4);
  float* coef = (float*)carve(24 * 4);

  const int n4 = HIDDEN * HIDDEN / 4;
  for (int i = 0; i < 4; ++i)
    pack_int8_kernel<<<(n4 + 255) / 256, 256, 0, stream>>>(
        w_i32[i], (unsigned char*)Wp[i], n4);

  quant_rows_kernel<<<NTOK / 8, 256, 0, stream>>>(hidden, Xq, xscale, NTOK);
  pla_coeff_kernel<<<1, 32, 0, stream>>>(coef);

  dim3 ggrid(HIDDEN / 64, NTOK / 128);
  for (int i = 0; i < 3; ++i)
    gemm_i8_kernel<0><<<ggrid, 256, 0, stream>>>(
        Xq, xscale, Wp[i], w_sc[i], w_bias[i], (void*)qkv[i]);

  dim3 agrid(SEQ / 64, NHEADS, BATCH);
  attn_kernel<<<agrid, 128, 0, stream>>>(qkv[0], qkv[1], qkv[2], mask, coef, ctx);

  quant_rows_kernel<<<NTOK / 8, 256, 0, stream>>>(ctx, Cq, cscale, NTOK);
  gemm_i8_kernel<1><<<ggrid, 256, 0, stream>>>(
      Cq, cscale, Wp[3], w_sc[3], w_bias[3], d_out);
}
